// FullAttention_37538014167158
// MI455X (gfx1250) — matrix operand order repaired and checked by reading, not compiled, not run on hardware
//
#include <hip/hip_runtime.h>

// ---------------------------------------------------------------------------
// Causal relative-position attention, flash-style single pass, CDNA5 WMMA.
//
// scores = (q+u_w)*scale . K^T  +  (q+v_w)*scale . P^T   (u_b/v_b are per-row
// constants -> cancel in softmax).  Computed transposed (S^T = K Q1^T+P Q2^T)
// so each lane owns one softmax row -> one shfl_xor(16) per reduction, and
// the probability tile feeds the O^T = V^T P^T WMMA B-operand with only a
// half-wave exchange.  Double-buffered LDS + register prefetch: one barrier
// per 32-key block, next tile's global loads overlap the 12 WMMAs.
// ---------------------------------------------------------------------------

typedef __attribute__((ext_vector_type(16))) _Float16 v16h;
typedef __attribute__((ext_vector_type(8)))  _Float16 v8h;
typedef __attribute__((ext_vector_type(8)))  float    v8f;
typedef __attribute__((ext_vector_type(4)))  float    v4f;

#define B_  8
#define L_  1024
#define H_  8
#define E_  64
#define S_L (H_ * E_)        // 512 floats between consecutive l (or s)
#define S_B (L_ * H_ * E_)   // floats per batch

union Frag16 { v16h v; v8h h[2]; unsigned u[8]; };
union U8h    { v8h  v; unsigned u[4]; };

static __device__ inline unsigned pk2(float a, float b) {
  auto r = __builtin_amdgcn_cvt_pkrtz(a, b);        // v_cvt_pk_rtz_f16_f32
  union { decltype(r) h; unsigned u; } t;
  t.h = r;
  return t.u;
}

__global__ __launch_bounds__(128)
void fullattn_rel_kernel(const float* __restrict__ q,
                         const float* __restrict__ k,
                         const float* __restrict__ v,
                         const float* __restrict__ p,
                         const float* __restrict__ uw,
                         const float* __restrict__ vw,
                         float* __restrict__ out)
{
  // Double-buffered tiles.  Row pads (72 / 40 halves) keep every per-lane
  // ds_load_b128 fragment read on distinct banks; strides stay 16B-aligned.
  __shared__ _Float16 Klds[2][32][72];   // [s][e]  f16
  __shared__ _Float16 Plds[2][32][72];   // [s][e]  f16
  __shared__ _Float16 Vlds[2][64][40];   // [e][s]  f16 (transposed)

  const int rowtile = blockIdx.x;        // 0..15 (64 query rows each)
  const int b    = blockIdx.y >> 3;
  const int h    = blockIdx.y & 7;
  const int tid  = threadIdx.x;
  const int wv   = tid >> 5;
  const int lane = tid & 31;
  const int n    = lane & 15;
  const bool hi  = lane >= 16;

  const int l_base = rowtile * 64 + wv * 16;
  const int lrow   = l_base + n;         // this lane's query row
  const float scale = 0.125f;            // 1/sqrt(E)
  const size_t gbase = (size_t)b * S_B + h * E_;

  // ---- Loop-invariant B fragments: Q1=(q+u_w)*s, Q2=(q+v_w)*s -------------
  // B 32x16 layout (K=e): lanes 0-15 hold K = eh*32+0..15, lanes 16-31 hold
  // K = eh*32+16..31; N = lane&15.
  const float* qrow = q + gbase + (size_t)lrow * S_L;
  Frag16 q1b[2], q2b[2];
#pragma unroll
  for (int eh = 0; eh < 2; ++eh) {
    const int e0 = eh * 32 + (hi ? 16 : 0);
#pragma unroll
    for (int c = 0; c < 4; ++c) {
      v4f qv = *(const v4f*)(qrow + e0 + c * 4);
      v4f uv = *(const v4f*)(uw + lrow * E_ + e0 + c * 4);
      v4f wv2 = *(const v4f*)(vw + lrow * E_ + e0 + c * 4);
      float f0 = (qv[0] + uv[0]) * scale, f1 = (qv[1] + uv[1]) * scale;
      float f2 = (qv[2] + uv[2]) * scale, f3 = (qv[3] + uv[3]) * scale;
      q1b[eh].u[c * 2]     = pk2(f0, f1);
      q1b[eh].u[c * 2 + 1] = pk2(f2, f3);
      float g0 = (qv[0] + wv2[0]) * scale, g1 = (qv[1] + wv2[1]) * scale;
      float g2 = (qv[2] + wv2[2]) * scale, g3 = (qv[3] + wv2[3]) * scale;
      q2b[eh].u[c * 2]     = pk2(g0, g1);
      q2b[eh].u[c * 2 + 1] = pk2(g2, g3);
    }
  }

  // ---- fill-stage thread mappings -----------------------------------------
  const int fs  = tid >> 2;            // K/P: row s 0..31
  const int fec = (tid & 3) * 16;      // K/P: e chunk
  const int ve  = tid & 63;            // V: lane -> e (coalesced b32 loads)
  const int vs  = (tid >> 6) * 16;     // V: s half 0 / 16

  U8h kreg[2], preg[2], vreg[2];       // prefetched tile (f16-packed)

  auto loadTile = [&](int s0) {
    const float* kp = k + gbase + (size_t)(s0 + fs) * S_L + fec;
    const float* pp = p + gbase + (size_t)(s0 + fs) * S_L + fec;
#pragma unroll
    for (int c = 0; c < 2; ++c) {
      v4f a0 = *(const v4f*)(kp + c * 8);
      v4f a1 = *(const v4f*)(kp + c * 8 + 4);
      v4f b0 = *(const v4f*)(pp + c * 8);
      v4f b1 = *(const v4f*)(pp + c * 8 + 4);
      kreg[c].u[0] = pk2(a0[0], a0[1]); kreg[c].u[1] = pk2(a0[2], a0[3]);
      kreg[c].u[2] = pk2(a1[0], a1[1]); kreg[c].u[3] = pk2(a1[2], a1[3]);
      preg[c].u[0] = pk2(b0[0], b0[1]); preg[c].u[1] = pk2(b0[2], b0[3]);
      preg[c].u[2] = pk2(b1[0], b1[1]); preg[c].u[3] = pk2(b1[2], b1[3]);
    }
    const float* vpp = v + gbase + ve + (size_t)(s0 + vs) * S_L;
    float vals[16];
#pragma unroll
    for (int i = 0; i < 16; ++i)
      vals[i] = vpp[(size_t)i * S_L];  // coalesced: lanes cover e 0..63
#pragma unroll
    for (int c = 0; c < 2; ++c)
#pragma unroll
      for (int j = 0; j < 4; ++j)
        vreg[c].u[j] = pk2(vals[c * 8 + 2 * j], vals[c * 8 + 2 * j + 1]);
  };

  auto storeTile = [&](int buf) {
    *(v8h*)&Klds[buf][fs][fec]     = kreg[0].v;
    *(v8h*)&Klds[buf][fs][fec + 8] = kreg[1].v;
    *(v8h*)&Plds[buf][fs][fec]     = preg[0].v;
    *(v8h*)&Plds[buf][fs][fec + 8] = preg[1].v;
    *(v8h*)&Vlds[buf][ve][vs]      = vreg[0].v;  // contiguous in s
    *(v8h*)&Vlds[buf][ve][vs + 8]  = vreg[1].v;
  };

  v8f Oacc[4] = {};                 // O^T tiles: e = t*16 + j (+8 hi lanes)
  float mrun = -1e30f, lrun = 0.0f; // online-softmax state per query row

  const int nblk = rowtile * 2 + 2; // causal: only s <= l_base+63 needed
  loadTile(0);

  for (int blk = 0; blk < nblk; ++blk) {
    const int s0  = blk * 32;
    const int buf = blk & 1;

    storeTile(buf);
    __syncthreads();                        // single barrier per block
    if (blk + 1 < nblk) loadTile((blk + 1) * 32);   // overlap with compute

    // Wave-uniform: block entirely above this wave's rows -> nothing to do.
    if (s0 <= l_base + 15) {
      // ---- S^T tiles (M = s, N = l): 8 WMMAs ----------------------------
      v8f Sacc[2] = {};
#pragma unroll
      for (int st = 0; st < 2; ++st) {
        const int srow = st * 16 + n;
        const int eb0  = hi ? 16 : 0;
#pragma unroll
        for (int eh = 0; eh < 2; ++eh) {
          const int eb = eh * 32 + eb0;
          Frag16 ka, pa;  // A 16x32 fragments (rows = s, K = e)
          ka.h[0] = *(const v8h*)&Klds[buf][srow][eb];
          ka.h[1] = *(const v8h*)&Klds[buf][srow][eb + 8];
          pa.h[0] = *(const v8h*)&Plds[buf][srow][eb];
          pa.h[1] = *(const v8h*)&Plds[buf][srow][eb + 8];
          Sacc[st] = __builtin_amdgcn_wmma_f32_16x16x32_f16(
              false, ka.v, false, q1b[eh].v, (short)0, Sacc[st], false, false);
          Sacc[st] = __builtin_amdgcn_wmma_f32_16x16x32_f16(
              false, pa.v, false, q2b[eh].v, (short)0, Sacc[st], false, false);
        }
      }

      // ---- causal mask: only diagonal blocks need the compares ----------
      if (s0 + 31 > l_base) {
#pragma unroll
        for (int j = 0; j < 8; ++j) {
          const int sg0 = s0 + j + (hi ? 8 : 0);
          if (sg0 > lrow)      Sacc[0][j] = -1e30f;
          if (sg0 + 16 > lrow) Sacc[1][j] = -1e30f;
        }
      }

      // ---- online softmax (lane owns row l = lrow) ----------------------
      float tmax = -1e30f;
#pragma unroll
      for (int j = 0; j < 8; ++j)
        tmax = fmaxf(tmax, fmaxf(Sacc[0][j], Sacc[1][j]));
      tmax = fmaxf(tmax, __shfl_xor(tmax, 16, 32));
      const float mnew = fmaxf(mrun, tmax);
      const float corr = __expf(mrun - mnew);
      float tsum = 0.0f;
#pragma unroll
      for (int j = 0; j < 8; ++j) {
        const float e0 = __expf(Sacc[0][j] - mnew);
        const float e1 = __expf(Sacc[1][j] - mnew);
        Sacc[0][j] = e0; Sacc[1][j] = e1;
        tsum += e0 + e1;
      }
      tsum += __shfl_xor(tsum, 16, 32);
      lrun = lrun * corr + tsum;
      mrun = mnew;
#pragma unroll
      for (int t = 0; t < 4; ++t)
#pragma unroll
        for (int j = 0; j < 8; ++j)
          Oacc[t][j] *= corr;

      // ---- pack probabilities into B fragment (32x16, K = s-offset) -----
      unsigned pk0[4], pk1[4];
#pragma unroll
      for (int c = 0; c < 4; ++c) {
        pk0[c] = pk2(Sacc[0][2 * c], Sacc[0][2 * c + 1]);
        pk1[c] = pk2(Sacc[1][2 * c], Sacc[1][2 * c + 1]);
      }
      Frag16 pb;
#pragma unroll
      for (int c = 0; c < 4; ++c) {
        const unsigned x0 = (unsigned)__shfl_xor((int)pk0[c], 16, 32);
        const unsigned x1 = (unsigned)__shfl_xor((int)pk1[c], 16, 32);
        pb.u[c]     = hi ? x1 : pk0[c];   // K 0..7  (lo) | 16..23 (hi)
        pb.u[4 + c] = hi ? pk1[c] : x0;   // K 8..15 (lo) | 24..31 (hi)
      }

      // ---- O^T += V^T * P^T : 4 WMMAs -----------------------------------
#pragma unroll
      for (int t = 0; t < 4; ++t) {
        const int erow = t * 16 + n;
        const int sb   = hi ? 16 : 0;
        Frag16 va;   // A 16x32 fragment (rows = e, K = s)
        va.h[0] = *(const v8h*)&Vlds[buf][erow][sb];
        va.h[1] = *(const v8h*)&Vlds[buf][erow][sb + 8];
        Oacc[t] = __builtin_amdgcn_wmma_f32_16x16x32_f16(
            false, va.v, false, pb.v, (short)0, Oacc[t], false, false);
      }
    }
  }

  // ---- normalize and store out[b, lrow, h, e] -----------------------------
  const float rinv = 1.0f / lrun;
  float* orow = out + gbase + (size_t)lrow * S_L;
#pragma unroll
  for (int t = 0; t < 4; ++t) {
    const int e0 = t * 16 + (hi ? 8 : 0);
    v4f w0, w1;
#pragma unroll
    for (int i = 0; i < 4; ++i) {
      w0[i] = Oacc[t][i] * rinv;
      w1[i] = Oacc[t][4 + i] * rinv;
    }
    *(v4f*)(orow + e0)     = w0;
    *(v4f*)(orow + e0 + 4) = w1;
  }
}

extern "C" void kernel_launch(void* const* d_in, const int* in_sizes, int n_in,
                              void* d_out, int out_size, void* d_ws, size_t ws_size,
                              hipStream_t stream) {
  // setup_inputs order: queries, keys, values, position_embedding_key,
  // attn_mask (unused: causality computed analytically), u_w,
  // u_b (unused: cancels in softmax), v_w, v_b (unused), d_keys (unused).
  const float* q  = (const float*)d_in[0];
  const float* k  = (const float*)d_in[1];
  const float* v  = (const float*)d_in[2];
  const float* p  = (const float*)d_in[3];
  const float* uw = (const float*)d_in[5];
  const float* vw = (const float*)d_in[7];
  float* out = (float*)d_out;

  dim3 grid(L_ / 64, B_ * H_);   // 16 row tiles x 64 (b,h) pairs
  fullattn_rel_kernel<<<grid, 128, 0, stream>>>(q, k, v, p, uw, vw, out);
}
